// Rwkv7SelfAttention_75316546503073
// MI455X (gfx1250) — matrix operand order repaired and checked by reading, not compiled, not run on hardware
//
#include <hip/hip_runtime.h>

// ---------------------------------------------------------------------------
// RWKV7 attention for MI455X (gfx1250, wave32, WMMA).
//   T=4096, HID=4096, H=64, HS=64.
//   Big GEMMs  -> bf16 WMMA (v_wmma_f32_16x16x32_bf16), f32 accumulate,
//                 LDS double-buffered via GLOBAL_LOAD_ASYNC_TO_LDS_B128
//                 (ASYNCcnt) when the toolchain exposes the builtin.
//   Scan       -> 1 wave per head, 64x64 state in VGPRs (2 rows/lane),
//                 LDS broadcast + __shfl_xor reductions (wave32 native).
// ---------------------------------------------------------------------------

typedef __bf16 bf16;
typedef __attribute__((ext_vector_type(8)))  __bf16 bf16x8;
typedef __attribute__((ext_vector_type(16))) __bf16 v16bf;
typedef __attribute__((ext_vector_type(8)))  float  v8f;
typedef int v4i __attribute__((vector_size(16)));

#define HID 4096
#define NH  64
#define HS  64

#define GLOBAL_AS __attribute__((address_space(1)))
#define LDS_AS    __attribute__((address_space(3)))

typedef GLOBAL_AS v4i* g_v4i_p;
typedef LDS_AS    v4i* l_v4i_p;

#if defined(__has_builtin)
#if __has_builtin(__builtin_amdgcn_global_load_async_to_lds_b128)
#define HAVE_ASYNC_LDS 1
#endif
#endif

#if defined(__has_builtin)
#if __has_builtin(__builtin_amdgcn_s_wait_asynccnt)
#define WAIT_ASYNC() __builtin_amdgcn_s_wait_asynccnt(0)
#else
#define WAIT_ASYNC() asm volatile("s_wait_asynccnt 0x0" ::: "memory")
#endif
#else
#define WAIT_ASYNC() asm volatile("s_wait_asynccnt 0x0" ::: "memory")
#endif

static __device__ __forceinline__ v16bf make_v16(bf16x8 lo, bf16x8 hi) {
  union { v16bf v; bf16x8 h[2]; } u;
  u.h[0] = lo; u.h[1] = hi;
  return u.v;
}

static __device__ __forceinline__ float sigmoidf_(float x) {
  return 1.0f / (1.0f + expf(-x));
}

static __device__ __forceinline__ float wave_sum(float x) {
#pragma unroll
  for (int off = 16; off >= 1; off >>= 1) x += __shfl_xor(x, off, 32);
  return x;
}

// ---------------------------------------------------------------------------
// fp32 -> bf16 elementwise convert (weights)
// ---------------------------------------------------------------------------
__global__ __launch_bounds__(256) void k_f2bf(const float* __restrict__ in,
                                              bf16* __restrict__ out, size_t n) {
  size_t i = (size_t)blockIdx.x * blockDim.x + threadIdx.x;
  size_t stride = (size_t)gridDim.x * blockDim.x;
  for (; i < n; i += stride) out[i] = (bf16)in[i];
}

// ---------------------------------------------------------------------------
// token shift + 6 mixes, output bf16
// ---------------------------------------------------------------------------
__global__ __launch_bounds__(256) void k_mix(
    const float* __restrict__ x, const float* __restrict__ state1,
    const float* __restrict__ mr, const float* __restrict__ mw,
    const float* __restrict__ mk, const float* __restrict__ mv,
    const float* __restrict__ ma, const float* __restrict__ mg,
    bf16* __restrict__ br, bf16* __restrict__ bw, bf16* __restrict__ bk,
    bf16* __restrict__ bv, bf16* __restrict__ ba, bf16* __restrict__ bg) {
  const int t = blockIdx.x;
  const float* cur  = x + (size_t)t * HID;
  const float* prev = (t == 0) ? state1 : (cur - HID);
  const size_t base = (size_t)t * HID;
  for (int c = threadIdx.x; c < HID; c += 256) {
    float xc = cur[c];
    float dx = prev[c] - xc;
    br[base + c] = (bf16)(xc + dx * mr[c]);
    bw[base + c] = (bf16)(xc + dx * mw[c]);
    bk[base + c] = (bf16)(xc + dx * mk[c]);
    bv[base + c] = (bf16)(xc + dx * mv[c]);
    ba[base + c] = (bf16)(xc + dx * ma[c]);
    bg[base + c] = (bf16)(xc + dx * mg[c]);
  }
}

// ---------------------------------------------------------------------------
// C[M,N] = A[M,K] @ B[N,K]^T   (bf16 in, fp32 out) via WMMA 16x16x32 bf16.
// Block: 256 thr = 8 waves (2x4), 128x128 C tile, K-slab 32 in LDS.
// Async path: double-buffered LDS slabs filled by GLOBAL_LOAD_ASYNC_TO_LDS.
// ---------------------------------------------------------------------------
#define TM 128
#define TN 128
#define TK 32

__global__ __launch_bounds__(256) void k_gemm_bf16_nt(
    const bf16* __restrict__ A, const bf16* __restrict__ B,
    float* __restrict__ C, int M, int N, int K) {
#if HAVE_ASYNC_LDS
  __shared__ __align__(16) bf16 As[2][TM][TK];   // 2 x 8KB
  __shared__ __align__(16) bf16 Bs[2][TM][TK];   // 2 x 8KB
#else
  __shared__ __align__(16) bf16 As[1][TM][TK];
  __shared__ __align__(16) bf16 Bs[1][TM][TK];
#endif

  const int tid  = threadIdx.x;
  const int wave = tid >> 5;
  const int lane = tid & 31;
  const int wm   = wave >> 2;   // 0..1  -> 64 rows each
  const int wn   = wave & 3;    // 0..3  -> 32 cols each
  const int bm   = blockIdx.y * TM;
  const int bn   = blockIdx.x * TN;
  const int lrow = lane & 15;
  const int half = lane >> 4;

  v8f acc[4][2];
#pragma unroll
  for (int i = 0; i < 4; ++i)
#pragma unroll
    for (int j = 0; j < 2; ++j)
      acc[i][j] = v8f{0.f, 0.f, 0.f, 0.f, 0.f, 0.f, 0.f, 0.f};

#if HAVE_ASYNC_LDS
  // -------- async double-buffered staging: memory -> LDS, no VGPR round-trip
  auto stage = [&](int buf, int k0) {
#pragma unroll
    for (int i = 0; i < 2; ++i) {
      int chunk = tid + i * 256;  // 0..511 16B chunks per slab
      int row = chunk >> 2;
      int cc  = (chunk & 3) * 8;
      const bf16* ga = &A[(size_t)(bm + row) * K + k0 + cc];
      const bf16* gb = &B[(size_t)(bn + row) * K + k0 + cc];
      __builtin_amdgcn_global_load_async_to_lds_b128(
          (g_v4i_p)(uintptr_t)(const void*)ga,
          (l_v4i_p)(LDS_AS void*)(void*)&As[buf][row][cc], 0, 0);
      __builtin_amdgcn_global_load_async_to_lds_b128(
          (g_v4i_p)(uintptr_t)(const void*)gb,
          (l_v4i_p)(LDS_AS void*)(void*)&Bs[buf][row][cc], 0, 0);
    }
  };
  stage(0, 0);
  WAIT_ASYNC();
  __syncthreads();
  int buf = 0;
  for (int k0 = 0; k0 < K; k0 += TK, buf ^= 1) {
    if (k0 + TK < K) stage(buf ^ 1, k0 + TK);   // overlap fill of next slab
#else
  const int buf = 0;
  for (int k0 = 0; k0 < K; k0 += TK) {
    // -------- sync staging fallback: global -> VGPR -> LDS
#pragma unroll
    for (int i = 0; i < 2; ++i) {
      int chunk = tid + i * 256;
      int row = chunk >> 2;
      int cc  = (chunk & 3) * 8;
      *(bf16x8*)&As[0][row][cc] = *(const bf16x8*)&A[(size_t)(bm + row) * K + k0 + cc];
      *(bf16x8*)&Bs[0][row][cc] = *(const bf16x8*)&B[(size_t)(bn + row) * K + k0 + cc];
    }
    if (k0 + TK < K) {
      __builtin_prefetch(&A[(size_t)(bm + (tid >> 1)) * K + k0 + TK], 0, 1);
      __builtin_prefetch(&B[(size_t)(bn + (tid >> 1)) * K + k0 + TK], 0, 1);
    }
    __syncthreads();
#endif

    // A fragments: lane half 0 -> K{0..7,16..23}; half 1 -> K{8..15,24..31}
    v16bf afrag[4];
#pragma unroll
    for (int i = 0; i < 4; ++i) {
      int r = wm * 64 + i * 16 + lrow;
      bf16x8 lo = *(const bf16x8*)&As[buf][r][half * 8];
      bf16x8 hi = *(const bf16x8*)&As[buf][r][16 + half * 8];
      afrag[i] = make_v16(lo, hi);
    }
    // B fragments: col N = lane&15; half 0 -> K{0..7,16..23}; half 1 -> K{8..15,24..31}
    v16bf bfrag[2];
#pragma unroll
    for (int j = 0; j < 2; ++j) {
      int r = wn * 32 + j * 16 + lrow;
      bf16x8 lo = *(const bf16x8*)&Bs[buf][r][half * 8];
      bf16x8 hi = *(const bf16x8*)&Bs[buf][r][16 + half * 8];
      bfrag[j] = make_v16(lo, hi);
    }
#pragma unroll
    for (int i = 0; i < 4; ++i)
#pragma unroll
      for (int j = 0; j < 2; ++j)
        acc[i][j] = __builtin_amdgcn_wmma_f32_16x16x32_bf16(
            false, afrag[i], false, bfrag[j], (short)0, acc[i][j], false, false);

#if HAVE_ASYNC_LDS
    WAIT_ASYNC();        // next slab fully landed in LDS
#endif
    __syncthreads();     // everyone done reading current slab
  }

  // C/D layout: VGPR p, lane L: row = p + (L>>4)*8, col = L&15
#pragma unroll
  for (int i = 0; i < 4; ++i)
#pragma unroll
    for (int j = 0; j < 2; ++j)
#pragma unroll
      for (int p = 0; p < 8; ++p) {
        int row = bm + wm * 64 + i * 16 + half * 8 + p;
        int col = bn + wn * 32 + j * 16 + lrow;
        C[(size_t)row * N + col] = acc[i][j][p];
      }
}

// ---------------------------------------------------------------------------
// thin MLP stage 1:  H[t,d] = act( sum_c Xb[t,c] * U[c,d] ),  d < D <= 256
// mode: 0 none, 1 tanh, 2 sigmoid
// ---------------------------------------------------------------------------
__global__ __launch_bounds__(256) void k_proj1(const bf16* __restrict__ Xb,
                                               const float* __restrict__ U,
                                               float* __restrict__ Hout,
                                               int D, int mode) {
  const int t = blockIdx.x;
  __shared__ float xrow[HID];
  for (int c = threadIdx.x; c < HID; c += 256)
    xrow[c] = (float)Xb[(size_t)t * HID + c];
  __syncthreads();
  for (int d = threadIdx.x; d < D; d += 256) {
    float acc = 0.f;
    for (int c = 0; c < HID; ++c) acc += xrow[c] * U[(size_t)c * D + d];
    if (mode == 1) acc = tanhf(acc);
    else if (mode == 2) acc = sigmoidf_(acc);
    Hout[(size_t)t * D + d] = acc;
  }
}

// ---------------------------------------------------------------------------
// thin MLP stage 2:  y[t,j] = sum_d Hin[t,d] * Vw[d,j]  then per-mode epilog
// mode 0: out = exp(-0.606531*sigmoid(y+bias))            (decay w)
// mode 1: out = sigmoid(y+bias)                           (a)
// mode 2: out = y                                         (g)
// mode 3: out = out + (vf - out)*sigmoid(y+bias)          (v residual mix)
// ---------------------------------------------------------------------------
__global__ __launch_bounds__(256) void k_proj2(const float* __restrict__ Hin,
                                               const float* __restrict__ Vw,
                                               const float* __restrict__ bias,
                                               float* __restrict__ out,
                                               const float* __restrict__ vf,
                                               int D, int mode) {
  const int t = blockIdx.x;
  __shared__ float hrow[256];
  for (int d = threadIdx.x; d < D; d += 256) hrow[d] = Hin[(size_t)t * D + d];
  __syncthreads();
  for (int j = threadIdx.x; j < HID; j += 256) {
    float acc = 0.f;
    for (int d = 0; d < D; ++d) acc += hrow[d] * Vw[(size_t)d * HID + j];
    const size_t idx = (size_t)t * HID + j;
    if (mode == 0) {
      out[idx] = expf(-0.606531f * sigmoidf_(acc + bias[j]));
    } else if (mode == 1) {
      out[idx] = sigmoidf_(acc + bias[j]);
    } else if (mode == 2) {
      out[idx] = acc;
    } else {
      float s  = sigmoidf_(acc + bias[j]);
      float vv = out[idx];
      out[idx] = vv + (vf[idx] - vv) * s;
    }
  }
}

// ---------------------------------------------------------------------------
// per (t,h) head prep: kk = l2norm(k*k_k); Ain=-kk; Bin=kk*a; k *= 1+(a-1)*k_a
// one wave per (t,h), 2 elems per lane, __shfl_xor norm reduction.
// ---------------------------------------------------------------------------
__global__ __launch_bounds__(32) void k_head_prep(
    float* __restrict__ K, const float* __restrict__ A,
    const float* __restrict__ kkw, const float* __restrict__ kaw,
    float* __restrict__ Ain, float* __restrict__ Bin) {
  const int b = blockIdx.x;
  const int t = b >> 6;
  const int h = b & 63;
  const int l = threadIdx.x;
  const size_t base = (size_t)t * HID + (size_t)h * HS;
  const int c0 = h * HS + l, c1 = c0 + 32;

  float k0 = K[base + l],      k1 = K[base + l + 32];
  float a0 = A[base + l],      a1 = A[base + l + 32];
  float q0 = k0 * kkw[c0],     q1 = k1 * kkw[c1];
  float ss = wave_sum(q0 * q0 + q1 * q1);
  float inv = 1.0f / (sqrtf(ss) + 1e-12f);
  q0 *= inv; q1 *= inv;
  Ain[base + l]      = -q0;
  Ain[base + l + 32] = -q1;
  Bin[base + l]      = q0 * a0;
  Bin[base + l + 32] = q1 * a1;
  K[base + l]        = k0 * (1.0f + (a0 - 1.0f) * kaw[c0]);
  K[base + l + 32]   = k1 * (1.0f + (a1 - 1.0f) * kaw[c1]);
}

// ---------------------------------------------------------------------------
// sequential wkv7 scan: 1 wave per head, state rows i=lane and i=lane+32
// fully register resident (128 VGPRs), LDS broadcast of per-step vectors,
// LN + bonus via wave32 __shfl_xor butterflies. Emits gated = (LN(out)+bonus)*g
// as bf16 for the final WMMA GEMM.
// ---------------------------------------------------------------------------
__global__ __launch_bounds__(32, 1) void k_scan(
    const float* __restrict__ R, const float* __restrict__ W,
    const float* __restrict__ Kh, const float* __restrict__ V,
    const float* __restrict__ Ain, const float* __restrict__ Bin,
    const float* __restrict__ G, const float* __restrict__ state2,
    const float* __restrict__ lnw, const float* __restrict__ lnb,
    const float* __restrict__ rkw, bf16* __restrict__ gated, int T) {
  const int h = blockIdx.x;
  const int l = threadIdx.x;
  const int i0 = l, i1 = l + 32;

  float S0[HS], S1[HS];
#pragma unroll
  for (int j = 0; j < HS; ++j) {
    S0[j] = state2[((size_t)h * HS + i0) * HS + j];
    S1[j] = state2[((size_t)h * HS + i1) * HS + j];
  }
  __shared__ float rs[HS], ws[HS], ks[HS], as2[HS], bs[HS];

  const float lw0 = lnw[h * HS + i0], lw1 = lnw[h * HS + i1];
  const float lb0 = lnb[h * HS + i0], lb1 = lnb[h * HS + i1];
  const float rk0 = rkw[h * HS + i0], rk1 = rkw[h * HS + i1];

  for (int t = 0; t < T; ++t) {
    const size_t base = (size_t)t * HID + (size_t)h * HS;
    float r0 = R[base + i0],   r1 = R[base + i1];
    float w0 = W[base + i0],   w1 = W[base + i1];
    float k0 = Kh[base + i0],  k1 = Kh[base + i1];
    float v0 = V[base + i0],   v1 = V[base + i1];
    float a0 = Ain[base + i0], a1 = Ain[base + i1];
    float b0 = Bin[base + i0], b1 = Bin[base + i1];
    rs[i0] = r0; rs[i1] = r1;
    ws[i0] = w0; ws[i1] = w1;
    ks[i0] = k0; ks[i1] = k1;
    as2[i0] = a0; as2[i1] = a1;
    bs[i0] = b0; bs[i1] = b1;
    __syncthreads();  // single-wave workgroup: barrier degrades to S_NOP

    float sa0 = 0.f, sa1 = 0.f;
#pragma unroll
    for (int j = 0; j < HS; ++j) {
      float aj = as2[j];
      sa0 += S0[j] * aj;
      sa1 += S1[j] * aj;
    }
    float o0 = 0.f, o1 = 0.f;
#pragma unroll
    for (int j = 0; j < HS; ++j) {
      float wj = ws[j], bj = bs[j], kj = ks[j], rj = rs[j];
      S0[j] = S0[j] * wj + sa0 * bj + v0 * kj;
      S1[j] = S1[j] * wj + sa1 * bj + v1 * kj;
      o0 += S0[j] * rj;
      o1 += S1[j] * rj;
    }
    // head-wide reductions: sum, sum of squares, bonus
    float ps = wave_sum(o0 + o1);
    float pq = wave_sum(o0 * o0 + o1 * o1);
    float pb = wave_sum(r0 * k0 * rk0 + r1 * k1 * rk1);
    float mu  = ps * (1.0f / HS);
    float var = pq * (1.0f / HS) - mu * mu;
    float inv = rsqrtf(var + 0.00064f);
    float x0 = (o0 - mu) * inv * lw0 + lb0 + pb * v0;
    float x1 = (o1 - mu) * inv * lw1 + lb1 + pb * v1;
    gated[base + i0] = (bf16)(x0 * G[base + i0]);
    gated[base + i1] = (bf16)(x1 * G[base + i1]);
    __syncthreads();
  }
}

// ---------------------------------------------------------------------------
// host launch
// ---------------------------------------------------------------------------
extern "C" void kernel_launch(void* const* d_in, const int* in_sizes, int n_in,
                              void* d_out, int out_size, void* d_ws, size_t ws_size,
                              hipStream_t stream) {
  (void)n_in; (void)out_size; (void)ws_size;
  const float* x      = (const float*)d_in[0];
  const float* state1 = (const float*)d_in[1];
  const float* state2 = (const float*)d_in[2];
  const float* vfirst = (const float*)d_in[3];
  const float* x_r = (const float*)d_in[4];
  const float* x_w = (const float*)d_in[5];
  const float* x_k = (const float*)d_in[6];
  const float* x_v = (const float*)d_in[7];
  const float* x_a = (const float*)d_in[8];
  const float* x_g = (const float*)d_in[9];
  const float* W_r = (const float*)d_in[10];
  const float* W_k = (const float*)d_in[11];
  const float* W_v = (const float*)d_in[12];
  const float* W_o = (const float*)d_in[13];
  const float* w1 = (const float*)d_in[14];
  const float* w2 = (const float*)d_in[15];
  const float* w0 = (const float*)d_in[16];
  const float* a1 = (const float*)d_in[17];
  const float* a2 = (const float*)d_in[18];
  const float* a0 = (const float*)d_in[19];
  const float* v1 = (const float*)d_in[20];
  const float* v2 = (const float*)d_in[21];
  const float* v0 = (const float*)d_in[22];
  const float* g1 = (const float*)d_in[23];
  const float* g2 = (const float*)d_in[24];
  const float* k_k = (const float*)d_in[25];
  const float* k_a = (const float*)d_in[26];
  const float* r_k = (const float*)d_in[27];
  const float* ln_w = (const float*)d_in[28];
  const float* ln_b = (const float*)d_in[29];

  const int T = in_sizes[0] / HID;
  const int DD = in_sizes[14] / HID;  // 128
  const int DA = in_sizes[17] / HID;  // 96
  const int DV = in_sizes[20] / HID;  // 64
  const int DG = in_sizes[23] / HID;  // 256

  char* ws = (char*)d_ws;
  const size_t MB = (size_t)1 << 20;
  // bf16 weights (32 MB each)
  bf16* wbr = (bf16*)(ws + 0 * MB);
  bf16* wbk = (bf16*)(ws + 32 * MB);
  bf16* wbv = (bf16*)(ws + 64 * MB);
  bf16* wbo = (bf16*)(ws + 96 * MB);
  // bf16 mixes (32 MB each)
  bf16* mbr = (bf16*)(ws + 128 * MB);
  bf16* mbk = (bf16*)(ws + 160 * MB);
  bf16* mbw = (bf16*)(ws + 192 * MB);
  bf16* mbv = (bf16*)(ws + 224 * MB);
  bf16* mba = (bf16*)(ws + 256 * MB);
  bf16* mbg = (bf16*)(ws + 288 * MB);
  // fp32 activations (64 MB each)
  float* rbuf = (float*)(ws + 320 * MB);
  float* kbuf = (float*)(ws + 384 * MB);
  float* vbuf = (float*)(ws + 448 * MB);
  float* wdec = (float*)(ws + 512 * MB);
  float* abuf = (float*)(ws + 576 * MB);
  float* gbuf = (float*)(ws + 640 * MB);
  // A_in / B_in alias dead regions (weights wbr/wbk and mixes mbr/mbk are
  // only needed through the r/k/v GEMMs, which complete before head_prep).
  float* ainb = (float*)(ws + 0 * MB);    // 64 MB over wbr+wbk
  float* binb = (float*)(ws + 128 * MB);  // 64 MB over mbr+mbk
  // gated bf16 GEMM input
  bf16* gated = (bf16*)(ws + 704 * MB);
  // thin MLP hiddens
  float* hw = (float*)(ws + 736 * MB);
  float* ha = (float*)(ws + 739 * MB);
  float* hv = (float*)(ws + 742 * MB);
  float* hg = (float*)(ws + 745 * MB);

  const size_t NW = (size_t)HID * HID;

  // 1) weights -> bf16
  k_f2bf<<<4096, 256, 0, stream>>>(W_r, wbr, NW);
  k_f2bf<<<4096, 256, 0, stream>>>(W_k, wbk, NW);
  k_f2bf<<<4096, 256, 0, stream>>>(W_v, wbv, NW);
  k_f2bf<<<4096, 256, 0, stream>>>(W_o, wbo, NW);

  // 2) token shift + mixes
  k_mix<<<T, 256, 0, stream>>>(x, state1, x_r, x_w, x_k, x_v, x_a, x_g,
                               mbr, mbw, mbk, mbv, mba, mbg);

  // 3) big WMMA GEMMs: r, k, v
  dim3 ggrid(HID / TN, T / TM);
  k_gemm_bf16_nt<<<ggrid, 256, 0, stream>>>(mbr, wbr, rbuf, T, HID, HID);
  k_gemm_bf16_nt<<<ggrid, 256, 0, stream>>>(mbk, wbk, kbuf, T, HID, HID);
  k_gemm_bf16_nt<<<ggrid, 256, 0, stream>>>(mbv, wbv, vbuf, T, HID, HID);

  // 4) decay path: w = exp(-0.606531*sigmoid(tanh(xw@w1)@w2 + w0))
  k_proj1<<<T, 256, 0, stream>>>(mbw, w1, hw, DD, 1);
  k_proj2<<<T, 256, 0, stream>>>(hw, w2, w0, wdec, nullptr, DD, 0);
  // 5) a path: a = sigmoid(xa@a1@a2 + a0)
  k_proj1<<<T, 256, 0, stream>>>(mba, a1, ha, DA, 0);
  k_proj2<<<T, 256, 0, stream>>>(ha, a2, a0, abuf, nullptr, DA, 1);
  // 6) v residual mix: v += (v_first - v)*sigmoid(xv@v1@v2 + v0)
  k_proj1<<<T, 256, 0, stream>>>(mbv, v1, hv, DV, 0);
  k_proj2<<<T, 256, 0, stream>>>(hv, v2, v0, vbuf, vfirst, DV, 3);
  // 7) gate: g = sigmoid(xg@g1)@g2
  k_proj1<<<T, 256, 0, stream>>>(mbg, g1, hg, DG, 2);
  k_proj2<<<T, 256, 0, stream>>>(hg, g2, nullptr, gbuf, nullptr, DG, 2);

  // 8) per-head prep: kk/l2norm, A_in, B_in, k update (in place)
  k_head_prep<<<T * NH, 32, 0, stream>>>(kbuf, abuf, k_k, k_a, ainb, binb);

  // 9) sequential scan (1 wave per head) -> gated bf16
  k_scan<<<NH, 32, 0, stream>>>(rbuf, wdec, kbuf, vbuf, ainb, binb, gbuf,
                                state2, ln_w, ln_b, r_k, gated, T);

  // 10) output projection: out = gated @ W_o^T  (fp32 out)
  k_gemm_bf16_nt<<<ggrid, 256, 0, stream>>>(gated, wbo, (float*)d_out, T, HID, HID);
}
